// ViTHybridBlock_24876450578617
// MI455X (gfx1250) — matrix operand order repaired and checked by reading, not compiled, not run on hardware
//
#include <hip/hip_runtime.h>
#include <hip/hip_bf16.h>
#include <math.h>

typedef __attribute__((ext_vector_type(16))) _Float16 v16h;
typedef __attribute__((ext_vector_type(8)))  _Float16 v8h;
typedef __attribute__((ext_vector_type(8)))  float    v8f;
typedef __attribute__((ext_vector_type(4)))  int      v4i;

#define B_  8
#define N_  1024
#define C_  512
#define NH_ 8
#define DH_ 64
#define FF_ 2048

__device__ __forceinline__ v8f wmma_ff(v16h a, v16h b, v8f c) {
    // D = A(16x32 f16) * B(32x16 f16) + C(16x16 f32)
    return __builtin_amdgcn_wmma_f32_16x16x32_f16(false, a, false, b, (short)0, c, false, false);
}

// A fragment 16x32: row-major A[lda], tile at (m0,k0).
// lane&15 = row; lane>>4 selects K chunk kg*16..kg*16+15
__device__ __forceinline__ v16h load_afrag(const _Float16* base, int lda, int m0, int k0) {
    int lane = threadIdx.x & 31;
    int m = m0 + (lane & 15);
    int kg = lane >> 4;
    const _Float16* p = base + (size_t)m * lda + k0 + kg * 16;
    union { v16h v; v8h h[2]; } u;
    u.h[0] = *(const v8h*)p;
    u.h[1] = *(const v8h*)(p + 8);
    return u.v;
}

// B fragment 32x16 where B[k][n] = W[n][k] (W row-major [Nc][K]).
// lane&15 = n; elements e: k = (lane>>4)*16 + e  (contiguous in W row)
__device__ __forceinline__ v16h load_bfrag_wt(const _Float16* W, int ldk, int n0, int k0) {
    int lane = threadIdx.x & 31;
    int n = n0 + (lane & 15);
    int kg = lane >> 4;
    const _Float16* p = W + (size_t)n * ldk + k0 + kg * 16;
    union { v16h v; v8h h[2]; } u;
    u.h[0] = *(const v8h*)p;
    u.h[1] = *(const v8h*)(p + 8);
    return u.v;
}

// ---- CDNA5 async global->LDS staging (ASYNCcnt path), with sync fallback ----
#if defined(__AMDGCN__) && __has_builtin(__builtin_amdgcn_global_load_async_to_lds_b128)
#define HAVE_ASYNC_LDS 1
typedef __attribute__((address_space(1))) v4i v4i_as1;
typedef __attribute__((address_space(3))) v4i v4i_as3;
#else
#define HAVE_ASYNC_LDS 0
#endif

__device__ __forceinline__ void async_copy16(const _Float16* g, _Float16* l) {
#if HAVE_ASYNC_LDS
    __builtin_amdgcn_global_load_async_to_lds_b128((v4i_as1*)g, (v4i_as3*)l, 0, 0);
#else
    *(v8h*)l = *(const v8h*)g;
#endif
}

__device__ __forceinline__ void async_join() {
#if HAVE_ASYNC_LDS
#if __has_builtin(__builtin_amdgcn_s_wait_asynccnt)
    __builtin_amdgcn_s_wait_asynccnt(0);
#else
    asm volatile("s_wait_asynccnt 0x0" ::: "memory");
#endif
#endif
}

// ---------------- f32 -> f16 convert ----------------
__global__ __launch_bounds__(256) void cvt_kernel(const float* __restrict__ src,
                                                  _Float16* __restrict__ dst, int n) {
    int i = blockIdx.x * 256 + threadIdx.x;
    if (i < n) dst[i] = (_Float16)src[i];
}

// ---------------- LayerNorm (f32 in, f16 out), one wave per 512-col row ----------------
__global__ __launch_bounds__(256) void ln_kernel(const float* __restrict__ xin,
                                                 const float* __restrict__ w,
                                                 const float* __restrict__ b,
                                                 _Float16* __restrict__ oh) {
    int wv = threadIdx.x >> 5, lane = threadIdx.x & 31;
    int row = blockIdx.x * 8 + wv;
    const float* xr = xin + (size_t)row * C_;
    float vals[16];
    float s = 0.f;
#pragma unroll
    for (int i = 0; i < 16; ++i) { float v = xr[lane + i * 32]; vals[i] = v; s += v; }
#pragma unroll
    for (int o = 1; o < 32; o <<= 1) s += __shfl_xor(s, o);
    float mu = s * (1.f / C_);
    float s2 = 0.f;
#pragma unroll
    for (int i = 0; i < 16; ++i) { float d = vals[i] - mu; s2 += d * d; }
#pragma unroll
    for (int o = 1; o < 32; o <<= 1) s2 += __shfl_xor(s2, o);
    float rs = rsqrtf(s2 * (1.f / C_) + 1e-6f);
    _Float16* orow = oh + (size_t)row * C_;
#pragma unroll
    for (int i = 0; i < 16; ++i) {
        int c = lane + i * 32;
        orow[c] = (_Float16)((vals[i] - mu) * rs * w[c] + b[c]);
    }
}

// ---------------- generic WMMA GEMM: Out[M,Nc] = A[M,K] @ W[Nc,K]^T ----------------
// Block = 8 waves, tile 128 rows x 64 cols; each wave does 16x64 (4 accumulators,
// A fragment reused 4x). Weight tile (64x32) staged through LDS once per block,
// via async global->LDS when available.
// MODE 0: scatter f16 to q/k/v [B,NH,N,DH]
// MODE 1: of[row,col] += val + bias[col]                      (of = x1, pre-filled with conv)
// MODE 2: oh[row,col] = f16(gelu_exact(val + bias[col]))
// MODE 3: of[row,col] = resid[row,col] + val + bias[col]      (final output)
template <int MODE>
__global__ __launch_bounds__(256)
void gemm16(const _Float16* __restrict__ A, const _Float16* __restrict__ W,
            const float* __restrict__ bias, const float* __restrict__ resid,
            float* of, _Float16* __restrict__ oh,
            _Float16* __restrict__ qd, _Float16* __restrict__ kd, _Float16* __restrict__ vd,
            int Nc, int K) {
    __shared__ _Float16 lds_b[64 * 32];       // [n_local][k] weight tile
    int tid = threadIdx.x;
    int w = tid >> 5;
    int lane = tid & 31;
    int m0 = blockIdx.y * 128 + w * 16;
    int n0blk = blockIdx.x * 64;
    int nn = lane & 15, hg = lane >> 4;
    int nl = tid >> 2, ch = tid & 3;          // staging role: row-of-W, 16B chunk
    const _Float16* wsrc = W + (size_t)(n0blk + nl) * K + ch * 8;
    _Float16* wdst = &lds_b[nl * 32 + ch * 8];

    v8f accs[4] = {};
    for (int k0 = 0; k0 < K; k0 += 32) {
        __syncthreads();                       // previous tile fully consumed
        async_copy16(wsrc + k0, wdst);         // 256 threads x 16B = 4KB tile
        async_join();
        if (k0 + 32 < K) {                     // prefetch next A tile (global_prefetch_b8)
            __builtin_prefetch(A + (size_t)(m0 + nn) * K + k0 + 32 + hg * 8, 0, 3);
        }
        __syncthreads();                       // tile visible to all waves
        v16h a = load_afrag(A, K, m0, k0);
#pragma unroll
        for (int t4 = 0; t4 < 4; ++t4) {
            union { v16h v; v8h h[2]; } u;
            const _Float16* bp = &lds_b[(t4 * 16 + nn) * 32 + hg * 16];
            u.h[0] = *(const v8h*)bp;
            u.h[1] = *(const v8h*)(bp + 8);
            accs[t4] = wmma_ff(a, u.v, accs[t4]);
        }
    }
#pragma unroll
    for (int t4 = 0; t4 < 4; ++t4) {
        int col = n0blk + t4 * 16 + nn;
#pragma unroll
        for (int r = 0; r < 8; ++r) {
            int row = m0 + r + 8 * hg;
            float val = accs[t4][r];
            if (MODE == 0) {
                int nsel = col >> 9, cc = col & 511, hh = cc >> 6, dd = cc & 63;
                int bb = row >> 10, tt = row & 1023;
                _Float16* dst = (nsel == 0) ? qd : ((nsel == 1) ? kd : vd);
                dst[(((size_t)bb * NH_ + hh) * N_ + tt) * DH_ + dd] = (_Float16)val;
            } else if (MODE == 1) {
                size_t idx = (size_t)row * Nc + col;
                of[idx] = of[idx] + val + bias[col];
            } else if (MODE == 2) {
                float t = val + bias[col];
                float g = 0.5f * t * (1.f + erff(t * 0.70710678118f));
                oh[(size_t)row * Nc + col] = (_Float16)g;
            } else {
                size_t idx = (size_t)row * Nc + col;
                of[idx] = resid[idx] + val + bias[col];
            }
        }
    }
}

// ---------------- top-k masks via mean-score trick + bitonic sort ----------------
// row_scores = (scale/N) * q_i . sum(k);  col_scores = (scale/N) * k_j . sum(q)
__global__ __launch_bounds__(256)
void mask_kernel(const _Float16* __restrict__ qh, const _Float16* __restrict__ kh,
                 float* __restrict__ rmask, float* __restrict__ cmask) {
    __shared__ float sv[64];
    __shared__ float part[256];
    __shared__ float scores[N_];
    __shared__ float srt[N_];
    int bid = blockIdx.x, bh = bid >> 1, which = bid & 1;
    const _Float16* sum_src = ((which == 0) ? kh : qh) + (size_t)bh * N_ * DH_;
    const _Float16* dot_src = ((which == 0) ? qh : kh) + (size_t)bh * N_ * DH_;
    int t = threadIdx.x;
    {   // sum over the 1024 rows of sum_src
        int d = t & 63, g = t >> 6;
        float s = 0.f;
        for (int j = g; j < N_; j += 4) s += (float)sum_src[(size_t)j * DH_ + d];
        part[t] = s;
    }
    __syncthreads();
    if (t < 64) sv[t] = part[t] + part[t + 64] + part[t + 128] + part[t + 192];
    __syncthreads();
    const float sc = 0.125f / (float)N_;   // DH^-0.5 / N
    for (int i = t; i < N_; i += 256) {
        float s = 0.f;
        const _Float16* r = dot_src + (size_t)i * DH_;
#pragma unroll
        for (int d = 0; d < DH_; ++d) s += (float)r[d] * sv[d];
        s *= sc;
        scores[i] = s; srt[i] = s;
    }
    __syncthreads();
    // bitonic sort, descending
    for (int k = 2; k <= N_; k <<= 1)
        for (int j = k >> 1; j > 0; j >>= 1) {
            for (int i = t; i < N_; i += 256) {
                int ixj = i ^ j;
                if (ixj > i) {
                    float a = srt[i], b = srt[ixj];
                    bool desc = ((i & k) == 0);
                    if (desc ? (a < b) : (a > b)) { srt[i] = b; srt[ixj] = a; }
                }
            }
            __syncthreads();
        }
    float thr = srt[511];   // HK = WK = 512 -> 512th largest
    float* msk = ((which == 0) ? rmask : cmask) + (size_t)bh * N_;
    for (int i = t; i < N_; i += 256) msk[i] = (scores[i] >= thr) ? 1.f : 0.f;
}

// ---------------- flash-attention with pre-softmax zero-masking ----------------
__global__ __launch_bounds__(256)
void attn_kernel(const _Float16* __restrict__ qh, const _Float16* __restrict__ kh,
                 const _Float16* __restrict__ vh, const float* __restrict__ rmask,
                 const float* __restrict__ cmask, _Float16* __restrict__ ao) {
    __shared__ _Float16 lds_vt[DH_ * 32];    // transposed V tile [vcol][jrow]
    __shared__ _Float16 lds_p[8][16 * 32];   // per-wave P tile
    int bh = blockIdx.x;
    int w = threadIdx.x >> 5, lane = threadIdx.x & 31;
    int r0 = blockIdx.y * 128 + w * 16;
    const _Float16* qb = qh + (size_t)bh * N_ * DH_;
    const _Float16* kb = kh + (size_t)bh * N_ * DH_;
    const _Float16* vb = vh + (size_t)bh * N_ * DH_;
    const float* rm = rmask + (size_t)bh * N_;
    const float* cm = cmask + (size_t)bh * N_;

    v16h qf0 = load_afrag(qb, DH_, r0, 0);
    v16h qf1 = load_afrag(qb, DH_, r0, 32);

    int hg = lane >> 4, nn = lane & 15;
    float rmr[8];
#pragma unroll
    for (int r = 0; r < 8; ++r) rmr[r] = rm[r0 + r + 8 * hg];

    float mrow[8], lrow[8];
#pragma unroll
    for (int r = 0; r < 8; ++r) { mrow[r] = -3.0e38f; lrow[r] = 0.f; }
    v8f acc0 = {}, acc1 = {}, acc2 = {}, acc3 = {};
    const float scale = 0.125f;   // DH^-0.5

    for (int j0 = 0; j0 < N_; j0 += 32) {
        __syncthreads();
        {   // stage V[j0..j0+31][0..63] transposed into LDS (shared by all waves)
            int jr = threadIdx.x >> 3, c0 = (threadIdx.x & 7) * 8;
            _Float16 tmp[8];
            *(v8h*)tmp = *(const v8h*)(vb + (size_t)(j0 + jr) * DH_ + c0);
#pragma unroll
            for (int e = 0; e < 8; ++e) lds_vt[(c0 + e) * 32 + jr] = tmp[e];
        }
        __syncthreads();
        // S tiles (16 rows x 32 cols) via two 16x16 WMMA accumulations each
        v8f s0 = {}, s1 = {};
        s0 = wmma_ff(qf0, load_bfrag_wt(kb, DH_, j0, 0), s0);
        s0 = wmma_ff(qf1, load_bfrag_wt(kb, DH_, j0, 32), s0);
        s1 = wmma_ff(qf0, load_bfrag_wt(kb, DH_, j0 + 16, 0), s1);
        s1 = wmma_ff(qf1, load_bfrag_wt(kb, DH_, j0 + 16, 32), s1);
        float cm0 = cm[j0 + nn], cm1 = cm[j0 + 16 + nn];
        float p0[8], p1[8], alpha[8];
#pragma unroll
        for (int r = 0; r < 8; ++r) {
            float a = s0[r] * scale * rmr[r] * cm0;   // masked entries -> exactly 0
            float b = s1[r] * scale * rmr[r] * cm1;
            float t = fmaxf(a, b);
#pragma unroll
            for (int o = 1; o < 16; o <<= 1) t = fmaxf(t, __shfl_xor(t, o));
            float mn = fmaxf(mrow[r], t);
            float al = expf(mrow[r] - mn);
            float e0 = expf(a - mn), e1 = expf(b - mn);
            float rs = e0 + e1;
#pragma unroll
            for (int o = 1; o < 16; o <<= 1) rs += __shfl_xor(rs, o);
            lrow[r] = lrow[r] * al + rs;
            mrow[r] = mn;
            alpha[r] = al; p0[r] = e0; p1[r] = e1;
        }
#pragma unroll
        for (int r = 0; r < 8; ++r) {
            acc0[r] *= alpha[r]; acc1[r] *= alpha[r];
            acc2[r] *= alpha[r]; acc3[r] *= alpha[r];
        }
        // bounce P (D-layout) through LDS into A-fragment layout
        _Float16* pw = &lds_p[w][0];
#pragma unroll
        for (int r = 0; r < 8; ++r) {
            pw[(r + 8 * hg) * 32 + nn]       = (_Float16)p0[r];
            pw[(r + 8 * hg) * 32 + 16 + nn]  = (_Float16)p1[r];
        }
        __syncthreads();
        v16h pf;
        {
            union { v16h v; v8h h[2]; } u;
            const _Float16* pp = pw + nn * 32 + hg * 16;
            u.h[0] = *(const v8h*)pp;
            u.h[1] = *(const v8h*)(pp + 8);
            pf = u.v;
        }
#pragma unroll
        for (int t4 = 0; t4 < 4; ++t4) {
            union { v16h v; v8h h[2]; } u;
            const _Float16* vp = &lds_vt[(t4 * 16 + nn) * 32 + hg * 16];
            u.h[0] = *(const v8h*)vp;
            u.h[1] = *(const v8h*)(vp + 8);
            if (t4 == 0)      acc0 = wmma_ff(pf, u.v, acc0);
            else if (t4 == 1) acc1 = wmma_ff(pf, u.v, acc1);
            else if (t4 == 2) acc2 = wmma_ff(pf, u.v, acc2);
            else              acc3 = wmma_ff(pf, u.v, acc3);
        }
    }
    // epilogue: normalize and write merged-heads layout [B,N,C] f16
    int b = bh >> 3, h = bh & 7;
#pragma unroll
    for (int r = 0; r < 8; ++r) {
        int row = r0 + r + 8 * hg;
        size_t obase = ((size_t)b * N_ + row) * C_ + h * DH_;
        float inv = 1.f / lrow[r];
        ao[obase +  0 + nn] = (_Float16)(acc0[r] * inv);
        ao[obase + 16 + nn] = (_Float16)(acc1[r] * inv);
        ao[obase + 32 + nn] = (_Float16)(acc2[r] * inv);
        ao[obase + 48 + nn] = (_Float16)(acc3[r] * inv);
    }
}

// ---------------- depthwise 3x3 conv (channel-last, coalesced) ----------------
__global__ __launch_bounds__(256)
void conv_kernel(const float* __restrict__ x, const float* __restrict__ cw,
                 const float* __restrict__ cb, float* __restrict__ x1) {
    int idx = blockIdx.x * 256 + threadIdx.x;   // B*N*C
    int c = idx & 511;
    int s = (idx >> 9) & 1023;
    int b = idx >> 19;
    int y = s >> 5, xc = s & 31;
    float acc = cb[c];
#pragma unroll
    for (int dy = -1; dy <= 1; ++dy) {
        int yy = y + dy;
        if (yy < 0 || yy > 31) continue;
#pragma unroll
        for (int dx = -1; dx <= 1; ++dx) {
            int xx = xc + dx;
            if (xx < 0 || xx > 31) continue;
            acc += x[((size_t)b * N_ + yy * 32 + xx) * C_ + c] * cw[c * 9 + (dy + 1) * 3 + (dx + 1)];
        }
    }
    x1[idx] = acc;
}

extern "C" void kernel_launch(void* const* d_in, const int* in_sizes, int n_in,
                              void* d_out, int out_size, void* d_ws, size_t ws_size,
                              hipStream_t stream) {
    (void)in_sizes; (void)n_in; (void)out_size; (void)ws_size;
    const float* x      = (const float*)d_in[0];
    const float* ln1_w  = (const float*)d_in[1];
    const float* ln1_b  = (const float*)d_in[2];
    const float* qkv_w  = (const float*)d_in[3];
    const float* out_w  = (const float*)d_in[4];
    const float* out_b  = (const float*)d_in[5];
    const float* conv_w = (const float*)d_in[6];
    const float* conv_b = (const float*)d_in[7];
    const float* ln2_w  = (const float*)d_in[8];
    const float* ln2_b  = (const float*)d_in[9];
    const float* ff1_w  = (const float*)d_in[10];
    const float* ff1_b  = (const float*)d_in[11];
    const float* ff2_w  = (const float*)d_in[12];
    const float* ff2_b  = (const float*)d_in[13];
    float* outp = (float*)d_out;

    const size_t M = (size_t)B_ * N_;   // 8192 rows
    char* ws = (char*)d_ws;
    auto alloc = [&](size_t bytes) { char* p = ws; ws += (bytes + 255) & ~(size_t)255; return p; };
    _Float16* xn_h   = (_Float16*)alloc(M * C_ * 2);
    _Float16* qkv_wh = (_Float16*)alloc((size_t)3 * C_ * C_ * 2);
    _Float16* out_wh = (_Float16*)alloc((size_t)C_ * C_ * 2);
    _Float16* ff1_wh = (_Float16*)alloc((size_t)FF_ * C_ * 2);
    _Float16* ff2_wh = (_Float16*)alloc((size_t)C_ * FF_ * 2);
    _Float16* q_h    = (_Float16*)alloc((size_t)B_ * NH_ * N_ * DH_ * 2);
    _Float16* k_h    = (_Float16*)alloc((size_t)B_ * NH_ * N_ * DH_ * 2);
    _Float16* v_h    = (_Float16*)alloc((size_t)B_ * NH_ * N_ * DH_ * 2);
    float*    rmask  = (float*)alloc((size_t)B_ * NH_ * N_ * 4);
    float*    cmask  = (float*)alloc((size_t)B_ * NH_ * N_ * 4);
    _Float16* ao_h   = (_Float16*)alloc(M * C_ * 2);
    float*    x1     = (float*)alloc(M * C_ * 4);
    _Float16* h1_h   = (_Float16*)alloc(M * C_ * 2);
    _Float16* h2_h   = (_Float16*)alloc(M * FF_ * 2);

    // weight conversions
    cvt_kernel<<<(3 * C_ * C_ + 255) / 256, 256, 0, stream>>>(qkv_w, qkv_wh, 3 * C_ * C_);
    cvt_kernel<<<(C_ * C_ + 255) / 256, 256, 0, stream>>>(out_w, out_wh, C_ * C_);
    cvt_kernel<<<(FF_ * C_ + 255) / 256, 256, 0, stream>>>(ff1_w, ff1_wh, FF_ * C_);
    cvt_kernel<<<(C_ * FF_ + 255) / 256, 256, 0, stream>>>(ff2_w, ff2_wh, C_ * FF_);

    // LN1 -> f16
    ln_kernel<<<M / 8, 256, 0, stream>>>(x, ln1_w, ln1_b, xn_h);

    // QKV projection, scatter to heads
    gemm16<0><<<dim3((3 * C_) / 64, M / 128), 256, 0, stream>>>(
        xn_h, qkv_wh, nullptr, nullptr, nullptr, nullptr, q_h, k_h, v_h, 3 * C_, C_);

    // top-k row/col masks (mean-score trick)
    mask_kernel<<<B_ * NH_ * 2, 256, 0, stream>>>(q_h, k_h, rmask, cmask);

    // flash attention
    attn_kernel<<<dim3(B_ * NH_, N_ / 128), 256, 0, stream>>>(q_h, k_h, v_h, rmask, cmask, ao_h);

    // conv branch into x1
    conv_kernel<<<(M * C_) / 256, 256, 0, stream>>>(x, conv_w, conv_b, x1);

    // out projection, accumulate into x1 (x_attn + x_conv)
    gemm16<1><<<dim3(C_ / 64, M / 128), 256, 0, stream>>>(
        ao_h, out_wh, out_b, nullptr, x1, nullptr, nullptr, nullptr, nullptr, C_, C_);

    // LN2 -> f16
    ln_kernel<<<M / 8, 256, 0, stream>>>(x1, ln2_w, ln2_b, h1_h);

    // FF1 + exact GELU -> f16
    gemm16<2><<<dim3(FF_ / 64, M / 128), 256, 0, stream>>>(
        h1_h, ff1_wh, ff1_b, nullptr, nullptr, h2_h, nullptr, nullptr, nullptr, FF_, C_);

    // FF2 + residual -> fp32 output
    gemm16<3><<<dim3(C_ / 64, M / 128), 256, 0, stream>>>(
        h2_h, ff2_wh, ff2_b, x1, outp, nullptr, nullptr, nullptr, nullptr, C_, FF_);
}
